// CAMIL_5050881540749
// MI455X (gfx1250) — matrix operand order repaired and checked by reading, not compiled, not run on hardware
//
#include <hip/hip_runtime.h>
#include <hip/hip_bf16.h>

// ---------------------------------------------------------------------------
// CAMIL pipeline for MI455X (gfx1250, wave32).
// Big GEMMs: f16 WMMA (v_wmma_f32_16x16x32_f16), LDS double-buffered 128x64
//   tiles; B stored transposed+padded so all fragment gathers are ds_load_b128;
//   next-tile global loads issued before current-tile WMMAs (latency hiding).
// Pinv Newton-Schulz 256^3 GEMMs: f32 WMMA (v_wmma_f32_16x16x4_f32).
// Everything else: VALU elementwise / LDS reductions / edge atomics.
// ---------------------------------------------------------------------------

typedef __attribute__((ext_vector_type(16))) _Float16 v16h;
typedef __attribute__((ext_vector_type(8)))  float    v8f;
typedef __attribute__((ext_vector_type(2)))  float    v2f;

#define NN    8192
#define DD    512
#define HH    8
#define DHD   64
#define MM    256
#define EE    131072
#define DQK2  256
#define DATT2 128
#define KCV   33

#define LDP   40   // padded LDS row stride in halfs (80B: b128-aligned, low conflict)

// ---------------------------------------------------------------------------
// f16-input / f32-accumulate tiled batched GEMM:  C = alpha * A(MxK) @ B(KxN)
// Block = 256 threads = 8 waves; block tile 128x64, K-step 32, double-buffered.
// Wave w: rows (w%4)*32, cols (w/4)*32 -> four 16x16 accumulators.
// All shapes used here are multiples of (128, 64, 32) -> no bounds checks.
// ---------------------------------------------------------------------------
__global__ __launch_bounds__(256) void gemm_f16w(
    const float* __restrict__ A, const float* __restrict__ B, float* __restrict__ C,
    int M, int N, int K, int lda, int ldb, int ldc,
    long long sA, long long sB, long long sC, float alpha)
{
  __shared__ _Float16 As[2][128 * LDP];   // [m][k] padded
  __shared__ _Float16 Bs[2][64 * LDP];    // [n][k] transposed + padded
  const int b = blockIdx.z;
  const float* Ab = A + (size_t)b * sA;
  const float* Bb = B + (size_t)b * sB;
  float* Cb = C + (size_t)b * sC;
  const int bm = blockIdx.y * 128;
  const int bn = blockIdx.x * 64;
  const int tid  = threadIdx.x;
  const int wave = tid >> 5;
  const int lane = tid & 31;
  const int wm   = (wave & 3) * 32;
  const int wn   = (wave >> 2) * 32;
  const int lrow = lane & 15;
  const int half = lane >> 4;

  // Per-thread staging coordinates (A: 128x32 as 4 float4; B: 32x64 as 2 float4).
  const int arow = tid >> 3;          // 0..31 (+32*i)
  const int acol = (tid & 7) * 4;     // 0..28
  const int brow = tid >> 4;          // 0..15 (+16*i)
  const int bcol = (tid & 15) * 4;    // 0..60

  v8f acc00 = {}, acc01 = {}, acc10 = {}, acc11 = {};
  float4 a4[4], b4[2];

  // Phase 1: issue ALL global loads for a K-tile (one clause, one wait).
  auto load_tile = [&](int kk) {
#pragma unroll
    for (int i = 0; i < 4; ++i)
      a4[i] = *(const float4*)(Ab + (size_t)(bm + arow + 32 * i) * lda + (kk + acol));
#pragma unroll
    for (int i = 0; i < 2; ++i)
      b4[i] = *(const float4*)(Bb + (size_t)(kk + brow + 16 * i) * ldb + (bn + bcol));
  };
  // Phase 2: convert f32->f16 and store to LDS buffer s (B transposed).
  auto store_tile = [&](int s) {
#pragma unroll
    for (int i = 0; i < 4; ++i) {
      _Float16* dst = &As[s][(arow + 32 * i) * LDP + acol];
      dst[0] = (_Float16)a4[i].x; dst[1] = (_Float16)a4[i].y;
      dst[2] = (_Float16)a4[i].z; dst[3] = (_Float16)a4[i].w;
    }
#pragma unroll
    for (int i = 0; i < 2; ++i) {
      int r = brow + 16 * i;
      Bs[s][(bcol + 0) * LDP + r] = (_Float16)b4[i].x;
      Bs[s][(bcol + 1) * LDP + r] = (_Float16)b4[i].y;
      Bs[s][(bcol + 2) * LDP + r] = (_Float16)b4[i].z;
      Bs[s][(bcol + 3) * LDP + r] = (_Float16)b4[i].w;
    }
  };

  load_tile(0);
  store_tile(0);
  __syncthreads();

  for (int k0 = 0; k0 < K; k0 += 32) {
    const int cur  = (k0 >> 5) & 1;
    const bool more = (k0 + 32) < K;
    if (more) load_tile(k0 + 32);              // in flight behind the WMMAs
    if (k0 + 64 < K) {                          // distance-2 cache hint
      __builtin_prefetch(Ab + (size_t)(bm + arow) * lda + (k0 + 64), 0, 1);
      __builtin_prefetch(Bb + (size_t)(k0 + 64 + brow) * ldb + bn, 0, 1);
    }

    // A/B fragments in contiguous k order (k = 16*half + e), same for both; all per-lane data contiguous -> ds_load_b128.
    v16h af0, af1;
    const _Float16* a0 = &As[cur][(wm + lrow) * LDP];
    const _Float16* a1 = &As[cur][(wm + 16 + lrow) * LDP];
#pragma unroll
    for (int e = 0; e < 16; ++e) {
      int v  = e >> 1;
      int kk = half * 16 + v * 2 + (e & 1);
      af0[e] = a0[kk];
      af1[e] = a1[kk];
    }
    v16h bf0, bf1;
    const _Float16* b0 = &Bs[cur][(wn + lrow) * LDP + half * 16];
    const _Float16* b1 = &Bs[cur][(wn + 16 + lrow) * LDP + half * 16];
#pragma unroll
    for (int e = 0; e < 16; ++e) { bf0[e] = b0[e]; bf1[e] = b1[e]; }

    acc00 = __builtin_amdgcn_wmma_f32_16x16x32_f16(false, af0, false, bf0, (short)0, acc00, false, false);
    acc01 = __builtin_amdgcn_wmma_f32_16x16x32_f16(false, af0, false, bf1, (short)0, acc01, false, false);
    acc10 = __builtin_amdgcn_wmma_f32_16x16x32_f16(false, af1, false, bf0, (short)0, acc10, false, false);
    acc11 = __builtin_amdgcn_wmma_f32_16x16x32_f16(false, af1, false, bf1, (short)0, acc11, false, false);

    if (more) store_tile(cur ^ 1);              // wait lands here, after WMMAs
    __syncthreads();
  }

  // C/D layout: VGPR v -> row v + 8*half, col lane%16.
#pragma unroll
  for (int v = 0; v < 8; ++v) {
    int m0 = bm + wm + half * 8 + v;
    int m1 = m0 + 16;
    int n0 = bn + wn + lrow;
    int n1 = n0 + 16;
    Cb[(size_t)m0 * ldc + n0] = alpha * acc00[v];
    Cb[(size_t)m0 * ldc + n1] = alpha * acc01[v];
    Cb[(size_t)m1 * ldc + n0] = alpha * acc10[v];
    Cb[(size_t)m1 * ldc + n1] = alpha * acc11[v];
  }
}

// ---------------------------------------------------------------------------
// f32 WMMA batched GEMM (one wave per 16x16 tile, K-step 8: loads for two
// WMMAs issued together) for the Newton-Schulz pseudoinverse chain.
// ---------------------------------------------------------------------------
__global__ __launch_bounds__(32) void gemm_f32w(
    const float* __restrict__ A, const float* __restrict__ B, float* __restrict__ C,
    int K, int lda, int ldb, int ldc,
    long long sA, long long sB, long long sC, float alpha)
{
  const int b = blockIdx.z;
  const float* Ab = A + (size_t)b * sA;
  const float* Bb = B + (size_t)b * sB;
  float* Cb = C + (size_t)b * sC;
  const int tm = blockIdx.y * 16;
  const int tn = blockIdx.x * 16;
  const int lane = threadIdx.x & 31;
  const int lrow = lane & 15;
  const int half = lane >> 4;
  const float* arow = Ab + (size_t)(tm + lrow) * lda;

  v8f acc = {};
  for (int k0 = 0; k0 < K; k0 += 8) {
    v2f a0, a1, b0, b1;
#pragma unroll
    for (int v = 0; v < 2; ++v) {
      a0[v] = arow[k0 + v + 2 * half];
      a1[v] = arow[k0 + 4 + v + 2 * half];
      b0[v] = Bb[(size_t)(k0 + v + 2 * half) * ldb + (tn + lrow)];
      b1[v] = Bb[(size_t)(k0 + 4 + v + 2 * half) * ldb + (tn + lrow)];
    }
    acc = __builtin_amdgcn_wmma_f32_16x16x4_f32(false, a0, false, b0, (short)0, acc, false, false);
    acc = __builtin_amdgcn_wmma_f32_16x16x4_f32(false, a1, false, b1, (short)0, acc, false, false);
  }
#pragma unroll
  for (int v = 0; v < 8; ++v)
    Cb[(size_t)(tm + half * 8 + v) * ldc + (tn + lrow)] = alpha * acc[v];
}

// ------------------------- support kernels ---------------------------------

__global__ void scale_q_k(float* qkv) {            // q *= DH^-0.5 (=1/8)
  int idx = blockIdx.x * blockDim.x + threadIdx.x;
  if (idx >= NN * DD) return;
  int col = idx & (DD - 1); int n = idx >> 9;
  qkv[(size_t)n * (3 * DD) + col] *= 0.125f;
}

__global__ void landmarks_k(const float* __restrict__ qkv, float* ql, float* kl) {
  int idx = blockIdx.x * blockDim.x + threadIdx.x;   // H*M*DH
  if (idx >= HH * MM * DHD) return;
  int d = idx & 63; int m = (idx >> 6) & 255; int h = idx >> 14;
  int col = h * 64 + d;
  float sq = 0.f, sk = 0.f;
  for (int j = 0; j < 32; ++j) {
    size_t rb = (size_t)(m * 32 + j) * (3 * DD);
    sq += qkv[rb + col];
    sk += qkv[rb + DD + col];
  }
  ql[idx] = sq * (1.f / 32.f);
  kl[idx] = sk * (1.f / 32.f);
}

__global__ void transpose_b_k(const float* __restrict__ in, float* out,
                              int Bn, int R, int C) {   // (B,R,C)->(B,C,R)
  int idx = blockIdx.x * blockDim.x + threadIdx.x;
  if (idx >= Bn * R * C) return;
  int c = idx % C; int t = idx / C; int r = t % R; int b = t / R;
  out[((size_t)b * C + c) * R + r] = in[idx];
}

__global__ void extract_kT_k(const float* __restrict__ qkv, float* kT) {
  int idx = blockIdx.x * blockDim.x + threadIdx.x;   // H*DH*N
  if (idx >= HH * DHD * NN) return;
  int n = idx & (NN - 1); int t = idx >> 13; int d = t & 63; int h = t >> 6;
  kT[idx] = qkv[(size_t)n * (3 * DD) + DD + h * 64 + d];
}

__global__ void softmax_rows_k(float* data, int L, int ld) {
  __shared__ float red[256];
  float* p = data + (size_t)blockIdx.x * ld;
  int t = threadIdx.x;
  float m = -1e30f;
  for (int i = t; i < L; i += 256) m = fmaxf(m, p[i]);
  red[t] = m; __syncthreads();
  for (int s = 128; s > 0; s >>= 1) { if (t < s) red[t] = fmaxf(red[t], red[t + s]); __syncthreads(); }
  m = red[0]; __syncthreads();
  float sum = 0.f;
  for (int i = t; i < L; i += 256) { float e = __expf(p[i] - m); p[i] = e; sum += e; }
  red[t] = sum; __syncthreads();
  for (int s = 128; s > 0; s >>= 1) { if (t < s) red[t] += red[t + s]; __syncthreads(); }
  float inv = 1.f / red[0];
  for (int i = t; i < L; i += 256) p[i] *= inv;
}

__global__ void colsum_k(const float* __restrict__ a2, float* cs) {
  int idx = blockIdx.x * blockDim.x + threadIdx.x;   // H*M
  if (idx >= HH * MM) return;
  int m = idx & 255; int h = idx >> 8;
  const float* p = a2 + ((size_t)h << 16) + m;
  float s = 0.f;
  for (int r = 0; r < MM; ++r) s += p[r * MM];
  cs[idx] = s;
}

__global__ void headmax_k(const float* __restrict__ cs, float* scale) {
  __shared__ float red[256];
  int t = threadIdx.x;
  red[t] = cs[blockIdx.x * 256 + t]; __syncthreads();
  for (int s = 128; s > 0; s >>= 1) { if (t < s) red[t] = fmaxf(red[t], red[t + s]); __syncthreads(); }
  if (t == 0) scale[blockIdx.x] = red[0];   // row sums of softmax == 1
}

__global__ void pinv_init_k(const float* __restrict__ a2, const float* __restrict__ scale,
                            float* z) {              // z0 = a2^T / scale
  int idx = blockIdx.x * blockDim.x + threadIdx.x;   // H*M*M
  if (idx >= HH * MM * MM) return;
  int j = idx & 255; int i = (idx >> 8) & 255; int h = idx >> 16;
  z[idx] = a2[((size_t)h << 16) + ((size_t)j << 8) + i] / scale[h];
}

__global__ void ci_minus_k(const float* __restrict__ X, float* Y, float c) {
  int idx = blockIdx.x * blockDim.x + threadIdx.x;   // H*M*M
  if (idx >= HH * MM * MM) return;
  int j = idx & 255; int i = (idx >> 8) & 255;
  Y[idx] = ((i == j) ? c : 0.f) - X[idx];
}

__global__ void conv_res_k(const float* __restrict__ qkv, const float* __restrict__ kern,
                           float* out) {             // depthwise conv along N, added
  int idx = blockIdx.x * blockDim.x + threadIdx.x;   // N*512 (n, h*64+d)
  if (idx >= NN * DD) return;
  int col = idx & (DD - 1); int n = idx >> 9; int h = col >> 6;
  const float* kh = kern + h * KCV;
  float s = 0.f;
  for (int t = 0; t < KCV; ++t) {
    int nn = n + t - KCV / 2;
    if (nn >= 0 && nn < NN) s += kh[t] * qkv[(size_t)nn * (3 * DD) + 2 * DD + col];
  }
  out[idx] += s;
}

__global__ void bias_res_k(const float* __restrict__ t, const float* __restrict__ bout,
                           const float* __restrict__ bag, float* enc) {
  int idx = blockIdx.x * blockDim.x + threadIdx.x;
  if (idx >= NN * DD) return;
  enc[idx] = t[idx] + bout[idx & (DD - 1)] + bag[idx];
}

__global__ void zero_k(float* p, int n) {
  int i = blockIdx.x * blockDim.x + threadIdx.x;
  if (i < n) p[i] = 0.f;
}

__global__ void edge_scores_k(const float* __restrict__ q2, const float* __restrict__ k2,
                              const int* __restrict__ er, const int* __restrict__ ec,
                              const float* __restrict__ adj, float* Araw) {
  int e = blockIdx.x * 8 + (threadIdx.x >> 5);
  int lane = threadIdx.x & 31;
  if (e >= EE) return;
  int r = er[e], c = ec[e];
  const float* qp = q2 + (size_t)r * DQK2;
  const float* kp = k2 + (size_t)c * DQK2;
  float s = 0.f;
#pragma unroll
  for (int j = 0; j < 8; ++j) s += qp[lane + 32 * j] * kp[lane + 32 * j];
#pragma unroll
  for (int off = 16; off > 0; off >>= 1) s += __shfl_xor(s, off, 32);
  if (lane == 0) atomicAdd(&Araw[r], adj[e] * s * (1.f / 16.f));
}

__global__ void softmax_vec_k(const float* __restrict__ in, float* out, int L) {
  __shared__ float red[1024];
  int t = threadIdx.x;
  float m = -1e30f;
  for (int i = t; i < L; i += 1024) m = fmaxf(m, in[i]);
  red[t] = m; __syncthreads();
  for (int s = 512; s > 0; s >>= 1) { if (t < s) red[t] = fmaxf(red[t], red[t + s]); __syncthreads(); }
  m = red[0]; __syncthreads();
  float sum = 0.f;
  for (int i = t; i < L; i += 1024) { float e = __expf(in[i] - m); out[i] = e; sum += e; }
  red[t] = sum; __syncthreads();
  for (int s = 512; s > 0; s >>= 1) { if (t < s) red[t] += red[t + s]; __syncthreads(); }
  float inv = 1.f / red[0];
  for (int i = t; i < L; i += 1024) out[i] *= inv;
}

__global__ void gate_k(const float* __restrict__ vt, const float* __restrict__ bv,
                       const float* __restrict__ na, const float* __restrict__ enc,
                       float* xo) {
  int idx = blockIdx.x * blockDim.x + threadIdx.x;
  if (idx >= NN * DD) return;
  int col = idx & (DD - 1); int n = idx >> 9;
  float val = vt[idx] + bv[col];
  float xl  = na[n] * val;
  float w   = 1.f / (1.f + __expf(xl));   // sigmoid(-xl)
  float sw  = w * w;
  xo[idx] = 2.f * xl * sw + 2.f * enc[idx] * (1.f - sw);
}

__global__ void att_comb_k(const float* __restrict__ ta, const float* __restrict__ tb,
                           float* att) {
  int idx = blockIdx.x * blockDim.x + threadIdx.x;
  if (idx >= NN * DATT2) return;
  att[idx] = tanhf(ta[idx]) * (1.f / (1.f + __expf(-tb[idx])));
}

__global__ void att_logits_k(const float* __restrict__ att, const float* __restrict__ w,
                             float* logits) {
  int row = blockIdx.x * 8 + (threadIdx.x >> 5);
  int lane = threadIdx.x & 31;
  if (row >= NN) return;
  const float* p = att + (size_t)row * DATT2;
  float s = 0.f;
#pragma unroll
  for (int j = 0; j < 4; ++j) s += p[lane + 32 * j] * w[lane + 32 * j];
#pragma unroll
  for (int off = 16; off > 0; off >>= 1) s += __shfl_xor(s, off, 32);
  if (lane == 0) logits[row] = s;
}

__global__ void pooled_k(const float* __restrict__ ka, const float* __restrict__ xo,
                         float* pooled) {
  int d = blockIdx.x * blockDim.x + threadIdx.x;
  if (d >= DD) return;
  float s = 0.f;
  for (int n = 0; n < NN; ++n) s += ka[n] * xo[(size_t)n * DD + d];
  pooled[d] = s;
}

__global__ void final_k(const float* __restrict__ pooled, const float* __restrict__ Wfc,
                        const float* __restrict__ bfc, float* out) {
  __shared__ float red[256];
  int t = threadIdx.x;
  red[t] = pooled[t] * Wfc[t] + pooled[t + 256] * Wfc[t + 256];
  __syncthreads();
  for (int s = 128; s > 0; s >>= 1) { if (t < s) red[t] += red[t + s]; __syncthreads(); }
  if (t == 0) out[0] = 1.f / (1.f + __expf(-(red[0] + bfc[0])));
}

// ------------------------------ driver -------------------------------------

extern "C" void kernel_launch(void* const* d_in, const int* in_sizes, int n_in,
                              void* d_out, int out_size, void* d_ws, size_t ws_size,
                              hipStream_t stream) {
  const float* bag   = (const float*)d_in[0];
  const int*   erow  = (const int*)  d_in[1];
  const int*   ecol  = (const int*)  d_in[2];
  const float* adj   = (const float*)d_in[3];
  const float* Wqkv  = (const float*)d_in[4];
  const float* Wout  = (const float*)d_in[5];
  const float* bout  = (const float*)d_in[6];
  const float* rker  = (const float*)d_in[7];
  const float* wq    = (const float*)d_in[8];
  const float* wk    = (const float*)d_in[9];
  const float* Wv    = (const float*)d_in[10];
  const float* bv    = (const float*)d_in[11];
  const float* v_att = (const float*)d_in[12];
  const float* u_att = (const float*)d_in[13];
  const float* w_att = (const float*)d_in[14];
  const float* Wfc   = (const float*)d_in[15];
  const float* bfc   = (const float*)d_in[16];

  float* outp    = (float*)d_out;        // [0] scalar
  float* A_raw   = outp + 1;             // [1..8192]
  float* k_alpha = outp + 1 + NN;        // [8193..16384]

  float* ws = (float*)d_ws;
  // ---- workspace layout (float offsets) ----
  constexpr size_t O_QKV = 0;                          // N*1536
  constexpr size_t O_S1  = O_QKV + (size_t)NN * 1536;  // H*N*M
  constexpr size_t O_S3  = O_S1 + (size_t)HH * NN * MM;// H*M*N
  constexpr size_t O_S2  = O_S3 + (size_t)HH * MM * NN;// H*M*M
  constexpr size_t O_ZA  = O_S2 + (size_t)HH * MM * MM;
  constexpr size_t O_ZB  = O_ZA + (size_t)HH * MM * MM;
  constexpr size_t O_P   = O_ZB + (size_t)HH * MM * MM;
  constexpr size_t O_Q   = O_P  + (size_t)HH * MM * MM;
  constexpr size_t O_R   = O_Q  + (size_t)HH * MM * MM;
  constexpr size_t O_QL  = O_R  + (size_t)HH * MM * MM;
  constexpr size_t O_KL  = O_QL + (size_t)HH * MM * DHD;
  constexpr size_t O_KLT = O_KL + (size_t)HH * MM * DHD;
  constexpr size_t O_KT  = O_KLT+ (size_t)HH * MM * DHD;
  constexpr size_t O_AV  = O_KT + (size_t)HH * DHD * NN;
  constexpr size_t O_ZAV = O_AV + (size_t)HH * MM * DHD;
  constexpr size_t O_NY  = O_ZAV+ (size_t)HH * MM * DHD;
  constexpr size_t O_T1  = O_NY + (size_t)NN * DD;
  constexpr size_t O_ENC = O_T1 + (size_t)NN * DD;
  constexpr size_t O_CS  = O_ENC+ (size_t)NN * DD;
  constexpr size_t O_SC  = O_CS + (size_t)HH * MM;
  constexpr size_t O_NA  = O_SC + HH;
  constexpr size_t O_LOG = O_NA + NN;
  constexpr size_t O_PO  = O_LOG+ NN;
  // stage-2 temporaries aliased over S1 (free after the a1 @ zav GEMM)
  constexpr size_t O_Q2  = O_S1;
  constexpr size_t O_K2  = O_S1 + (size_t)NN * DQK2;
  constexpr size_t O_VT  = O_K2 + (size_t)NN * DQK2;
  constexpr size_t O_XO  = O_VT + (size_t)NN * DD;
  constexpr size_t O_TA  = O_XO + (size_t)NN * DD;
  constexpr size_t O_TB  = O_TA + (size_t)NN * DATT2;
  constexpr size_t O_ATT = O_TB + (size_t)NN * DATT2;
  (void)in_sizes; (void)n_in; (void)out_size; (void)ws_size; (void)O_PO;

  auto gemm16 = [&](const float* A, const float* B, float* C, int M, int N, int K,
                    int lda, int ldb, int ldc, long long sA, long long sB, long long sC,
                    int batch, float alpha) {
    dim3 g(N / 64, M / 128, batch);
    gemm_f16w<<<g, 256, 0, stream>>>(A, B, C, M, N, K, lda, ldb, ldc, sA, sB, sC, alpha);
  };
  auto gemm32 = [&](const float* A, const float* B, float* C, float alpha) {
    dim3 g(MM / 16, MM / 16, HH);   // 256x256x256, batch=H
    gemm_f32w<<<g, 32, 0, stream>>>(A, B, C, MM, MM, MM, MM,
                                    (long long)MM * MM, (long long)MM * MM,
                                    (long long)MM * MM, alpha);
  };
  auto ew = [&](int n) { return dim3((n + 255) / 256); };

  // ================= stage 1: Nystrom encoder =================
  gemm16(bag, Wqkv, ws + O_QKV, NN, 3 * DD, DD, DD, 3 * DD, 3 * DD, 0, 0, 0, 1, 1.f);
  scale_q_k<<<ew(NN * DD), 256, 0, stream>>>(ws + O_QKV);
  landmarks_k<<<ew(HH * MM * DHD), 256, 0, stream>>>(ws + O_QKV, ws + O_QL, ws + O_KL);
  transpose_b_k<<<ew(HH * MM * DHD), 256, 0, stream>>>(ws + O_KL, ws + O_KLT, HH, MM, DHD);
  // a1 = softmax(q @ kl^T): batched over heads via 64-col strides into qkv
  gemm16(ws + O_QKV, ws + O_KLT, ws + O_S1, NN, MM, DHD,
         3 * DD, MM, MM, 64, (long long)DHD * MM, (long long)NN * MM, HH, 1.f);
  softmax_rows_k<<<HH * NN, 256, 0, stream>>>(ws + O_S1, MM, MM);
  // a2 = softmax(ql @ kl^T)
  gemm16(ws + O_QL, ws + O_KLT, ws + O_S2, MM, MM, DHD,
         DHD, MM, MM, (long long)MM * DHD, (long long)DHD * MM, (long long)MM * MM, HH, 1.f);
  softmax_rows_k<<<HH * MM, 256, 0, stream>>>(ws + O_S2, MM, MM);
  // a3 = softmax(ql @ k^T)
  extract_kT_k<<<ew(HH * DHD * NN), 256, 0, stream>>>(ws + O_QKV, ws + O_KT);
  gemm16(ws + O_QL, ws + O_KT, ws + O_S3, MM, NN, DHD,
         DHD, NN, NN, (long long)MM * DHD, (long long)DHD * NN, (long long)MM * NN, HH, 1.f);
  softmax_rows_k<<<HH * MM, 256, 0, stream>>>(ws + O_S3, NN, NN);
  // pinv(a2): z0 = a2^T / (max col-sum), 6 Newton-Schulz iterations (f32 WMMA)
  colsum_k<<<ew(HH * MM), 256, 0, stream>>>(ws + O_S2, ws + O_CS);
  headmax_k<<<HH, 256, 0, stream>>>(ws + O_CS, ws + O_SC);
  pinv_init_k<<<ew(HH * MM * MM), 256, 0, stream>>>(ws + O_S2, ws + O_SC, ws + O_ZA);
  float* zcur = ws + O_ZA; float* znxt = ws + O_ZB;
  for (int it = 0; it < 6; ++it) {
    gemm32(ws + O_S2, zcur, ws + O_P, 1.f);                               // xz
    ci_minus_k<<<ew(HH * MM * MM), 256, 0, stream>>>(ws + O_P, ws + O_Q, 7.f);
    gemm32(ws + O_P, ws + O_Q, ws + O_R, 1.f);                            // xz(7I-xz)
    ci_minus_k<<<ew(HH * MM * MM), 256, 0, stream>>>(ws + O_R, ws + O_Q, 15.f);
    gemm32(ws + O_P, ws + O_Q, ws + O_R, 1.f);                            // xz(15I-..)
    ci_minus_k<<<ew(HH * MM * MM), 256, 0, stream>>>(ws + O_R, ws + O_Q, 13.f);
    gemm32(zcur, ws + O_Q, znxt, 0.25f);                                  // z update
    float* t = zcur; zcur = znxt; znxt = t;
  }
  // av = a3 @ v ; zav = pinv @ av ; nyout = a1 @ zav  (+ depthwise conv residual)
  gemm16(ws + O_S3, ws + O_QKV + 2 * DD, ws + O_AV, MM, DHD, NN,
         NN, 3 * DD, DHD, (long long)MM * NN, 64, (long long)MM * DHD, HH, 1.f);
  gemm16(zcur, ws + O_AV, ws + O_ZAV, MM, DHD, MM,
         MM, DHD, DHD, (long long)MM * MM, (long long)MM * DHD, (long long)MM * DHD, HH, 1.f);
  gemm16(ws + O_S1, ws + O_ZAV, ws + O_NY, NN, DHD, MM,
         MM, DHD, DD, (long long)NN * MM, (long long)MM * DHD, 64, HH, 1.f);
  conv_res_k<<<ew(NN * DD), 256, 0, stream>>>(ws + O_QKV, rker, ws + O_NY);
  gemm16(ws + O_NY, Wout, ws + O_T1, NN, DD, DD, DD, DD, DD, 0, 0, 0, 1, 1.f);
  bias_res_k<<<ew(NN * DD), 256, 0, stream>>>(ws + O_T1, bout, bag, ws + O_ENC);

  // ================= stage 2: sparse edge attention =================
  gemm16(ws + O_ENC, wq, ws + O_Q2, NN, DQK2, DD, DD, DQK2, DQK2, 0, 0, 0, 1, 1.f);
  gemm16(ws + O_ENC, wk, ws + O_K2, NN, DQK2, DD, DD, DQK2, DQK2, 0, 0, 0, 1, 1.f);
  zero_k<<<ew(NN), 256, 0, stream>>>(A_raw, NN);
  edge_scores_k<<<EE / 8, 256, 0, stream>>>(ws + O_Q2, ws + O_K2, erow, ecol, adj, A_raw);
  softmax_vec_k<<<1, 1024, 0, stream>>>(A_raw, ws + O_NA, NN);
  gemm16(bag, Wv, ws + O_VT, NN, DD, DD, DD, DD, DD, 0, 0, 0, 1, 1.f);
  gate_k<<<ew(NN * DD), 256, 0, stream>>>(ws + O_VT, bv, ws + O_NA, ws + O_ENC, ws + O_XO);

  // ================= stage 3: gated MIL head =================
  gemm16(ws + O_XO, v_att, ws + O_TA, NN, DATT2, DD, DD, DATT2, DATT2, 0, 0, 0, 1, 1.f);
  gemm16(ws + O_XO, u_att, ws + O_TB, NN, DATT2, DD, DD, DATT2, DATT2, 0, 0, 0, 1, 1.f);
  att_comb_k<<<ew(NN * DATT2), 256, 0, stream>>>(ws + O_TA, ws + O_TB, ws + O_ATT);
  att_logits_k<<<NN / 8, 256, 0, stream>>>(ws + O_ATT, w_att, ws + O_LOG);
  softmax_vec_k<<<1, 1024, 0, stream>>>(ws + O_LOG, k_alpha, NN);
  pooled_k<<<2, 256, 0, stream>>>(k_alpha, ws + O_XO, ws + O_PO);
  final_k<<<1, 256, 0, stream>>>(ws + O_PO, Wfc, bfc, outp);
}
